// CustomModel_74586402063130
// MI455X (gfx1250) — compile-verified, matrix operand order checked by reading, not run on hardware
//
#include <hip/hip_runtime.h>
#include <hip/hip_bf16.h>

typedef __attribute__((ext_vector_type(16))) _Float16 v16h;
typedef __attribute__((ext_vector_type(8)))  _Float16 v8h;
typedef __attribute__((ext_vector_type(8)))  float    v8f;

#define TT 256      // sequence length
#define VV 10000    // vocab
#define ED 300      // embedding dim
#define MD 128      // midi dim
#define HH 256      // hidden
#define BT 8192     // B*T rows
#define H3 768      // 3*H
#define EP 320      // E padded to multiple of 32
#define HPAD 264    // LDS h row stride (halves), bank-conflict pad
#define GPAD 776    // LDS gh row stride (floats), bank-conflict pad

__device__ __forceinline__ v8f wmma16(v16h a, v16h b, v8f c) {
  return __builtin_amdgcn_wmma_f32_16x16x32_f16(false, a, false, b, (short)0, c, false, false);
}

// Load one 16x32 f16 A-tile from a row-major matrix (stride in halves).
// ISA 7.12.2 16-bit A layout: lanes 0-15: M=lane, K = {k0..k0+7, k0+16..k0+23}
//                             lanes 16-31: M=lane-16, K = {k0+8..k0+15, k0+24..k0+31}
__device__ __forceinline__ v16h load_a16(const _Float16* base, int stride, int lane) {
  const int m = lane & 15;
  const _Float16* p = base + (size_t)m * stride + ((lane < 16) ? 0 : 8);
  v8h c0 = *(const v8h*)(p);
  v8h c1 = *(const v8h*)(p + 16);
  v16h a;
#pragma unroll
  for (int i = 0; i < 8; ++i) { a[i] = c0[i]; a[i + 8] = c1[i]; }
  return a;
}

// B tiles are pre-swizzled: 512 halves per tile, lane-major (lane*16 contiguous).
__device__ __forceinline__ v16h load_b16(const _Float16* tile, int lane) {
  return *(const v16h*)(tile + lane * 16);
}

// Pack row-major f32 [K x N] weight into per-tile swizzled f16 B-operand layout.
__global__ __launch_bounds__(256) void k_pack_b(const float* __restrict__ src,
                                                _Float16* __restrict__ dst,
                                                int K, int N, int nTiles, int total) {
  int p = blockIdx.x * 256 + threadIdx.x;
  if (p >= total) return;
  int tile = p >> 9, q = p & 511;
  int lane = q >> 4, idx = q & 15;
  int kt = tile / nTiles, nt = tile - kt * nTiles;
  int kk = (lane < 16) ? ((idx < 8) ? idx : idx + 8)
                       : ((idx < 8) ? idx + 8 : idx + 16);
  int k = kt * 32 + kk;
  int n = nt * 16 + (lane & 15);
  float v = (k < K && n < N) ? src[(size_t)k * N + n] : 0.f;
  dst[p] = (_Float16)v;
}

__global__ __launch_bounds__(256) void k_f32_to_f16(const float* __restrict__ src,
                                                    _Float16* __restrict__ dst, int total) {
  int i = blockIdx.x * 256 + threadIdx.x;
  if (i < total) dst[i] = (_Float16)src[i];
}

// x = midis @ Wm + bm + E[words]; store f16 padded to EP columns (zeros beyond ED).
// 4 N-tiles per wave: A loaded once per K-step.
__global__ __launch_bounds__(256) void k1_embed(const _Float16* __restrict__ midis_h,
                                                const _Float16* __restrict__ Wm_swz,
                                                const int* __restrict__ words,
                                                const float* __restrict__ Emb,
                                                const float* __restrict__ bm,
                                                _Float16* __restrict__ x_h) {
  const int lane = threadIdx.x & 31;
  const int g = blockIdx.x * 8 + (threadIdx.x >> 5);      // 512*5 groups
  const int mt = g / 5, nt0 = (g - mt * 5) * 4;
  const int m0 = mt * 16;
  v8f acc[4] = {};
#pragma unroll
  for (int kt = 0; kt < 4; ++kt) {
    v16h a = load_a16(midis_h + (size_t)m0 * MD + kt * 32, MD, lane);
#pragma unroll
    for (int j = 0; j < 4; ++j) {
      v16h b = load_b16(Wm_swz + ((size_t)kt * 20 + nt0 + j) * 512, lane);
      acc[j] = wmma16(a, b, acc[j]);
    }
  }
  const int mBase = m0 + ((lane < 16) ? 0 : 8);
#pragma unroll
  for (int j = 0; j < 4; ++j) {
    const int n = (nt0 + j) * 16 + (lane & 15);
#pragma unroll
    for (int r = 0; r < 8; ++r) {
      int m = mBase + r;
      float v = acc[j][r];
      if (n < ED) v += bm[n] + Emb[(size_t)words[m] * ED + n];
      x_h[(size_t)m * EP + n] = (_Float16)v;
    }
  }
}

// gx = x @ Wx + bx   ([8192,320pad] x [320,768] -> f32), 4 N-tiles per wave.
__global__ __launch_bounds__(256) void k2_gx(const _Float16* __restrict__ x_h,
                                             const _Float16* __restrict__ Wx_swz,
                                             const float* __restrict__ bx,
                                             float* __restrict__ gx) {
  const int lane = threadIdx.x & 31;
  const int g = blockIdx.x * 8 + (threadIdx.x >> 5);      // 512*12 groups
  const int mt = g / 12, nt0 = (g - mt * 12) * 4;
  const int m0 = mt * 16;
  v8f acc[4] = {};
#pragma unroll
  for (int kt = 0; kt < 10; ++kt) {
    v16h a = load_a16(x_h + (size_t)m0 * EP + kt * 32, EP, lane);
#pragma unroll
    for (int j = 0; j < 4; ++j) {
      v16h b = load_b16(Wx_swz + ((size_t)kt * 48 + nt0 + j) * 512, lane);
      acc[j] = wmma16(a, b, acc[j]);
    }
  }
  const int mBase = m0 + ((lane < 16) ? 0 : 8);
#pragma unroll
  for (int j = 0; j < 4; ++j) {
    const int n = (nt0 + j) * 16 + (lane & 15);
    const float bias = bx[n];
#pragma unroll
    for (int r = 0; r < 8; ++r)
      gx[(size_t)(mBase + r) * H3 + n] = acc[j][r] + bias;
  }
}

// Sequential GRU scan. 2 workgroups x 16 batch rows; h lives in LDS.
// Each wave owns 2 N-tiles; A-tile (h strip) loaded once per K-step.
__global__ __launch_bounds__(768) void k3_gru(const float* __restrict__ gx,
                                              const _Float16* __restrict__ U_swz,
                                              const float* __restrict__ bu,
                                              const int* __restrict__ words,
                                              _Float16* __restrict__ seq_h) {
  __shared__ __align__(32) _Float16 h[16 * HPAD];
  __shared__ float gh[16 * GPAD];
  const int tid = threadIdx.x;
  const int lane = tid & 31, wave = tid >> 5;   // 24 waves
  const int bBase = blockIdx.x * 16;
  const int nt0 = wave * 2;
  for (int i = tid; i < 16 * HPAD; i += 768) h[i] = (_Float16)0.f;
  __syncthreads();

  for (int t = 0; t < TT; ++t) {
    // gh = h @ U + bu : 48 n-tiles, 2 per wave, K = 256 (8 WMMA steps)
    v8f acc0 = {}, acc1 = {};
#pragma unroll
    for (int kt = 0; kt < 8; ++kt) {
      v16h a = load_a16(h + kt * 32, HPAD, lane);
      v16h b0 = load_b16(U_swz + ((size_t)kt * 48 + nt0) * 512, lane);
      v16h b1 = load_b16(U_swz + ((size_t)kt * 48 + nt0 + 1) * 512, lane);
      acc0 = wmma16(a, b0, acc0);
      acc1 = wmma16(a, b1, acc1);
    }
    {
      const int mBase = (lane < 16) ? 0 : 8;
      const int n0 = nt0 * 16 + (lane & 15);
      const int n1 = n0 + 16;
      const float bias0 = bu[n0], bias1 = bu[n1];
#pragma unroll
      for (int r = 0; r < 8; ++r) {
        gh[(mBase + r) * GPAD + n0] = acc0[r] + bias0;
        gh[(mBase + r) * GPAD + n1] = acc1[r] + bias1;
      }
    }
    __syncthreads();

    // Gate math + masked carry (Keras reset_after GRU)
    for (int i = tid; i < 16 * HH; i += 768) {
      const int bl = i >> 8, j = i & 255;
      const int row = (bBase + bl) * TT + t;
      const float hz = gh[bl * GPAD + j];
      const float hr = gh[bl * GPAD + HH + j];
      const float hn = gh[bl * GPAD + 2 * HH + j];
      const float xz = gx[(size_t)row * H3 + j];
      const float xr = gx[(size_t)row * H3 + HH + j];
      const float xn = gx[(size_t)row * H3 + 2 * HH + j];
      const float z = 1.f / (1.f + __expf(-(xz + hz)));
      const float r = 1.f / (1.f + __expf(-(xr + hr)));
      const float n = tanhf(xn + r * hn);
      const float hp = (float)h[bl * HPAD + j];
      float hnew = z * hp + (1.f - z) * n;
      if (words[row] == 0) hnew = hp;        // mask: carry state through pads
      h[bl * HPAD + j] = (_Float16)hnew;
      seq_h[(size_t)row * HH + j] = (_Float16)hnew;
    }
    __syncthreads();
  }
}

// logits = seq @ Wo + bo   ([8192,256] x [256,10000] -> f32 out)
// 5 N-tiles per wave: A loaded once per K-step, 5 WMMAs per A load.
__global__ __launch_bounds__(256) void k4_out(const _Float16* __restrict__ seq_h,
                                              const _Float16* __restrict__ Wo_swz,
                                              const float* __restrict__ bo,
                                              float* __restrict__ out) {
  const int lane = threadIdx.x & 31;
  const int g = blockIdx.x * 8 + (threadIdx.x >> 5);      // 512*125 groups
  const int mt = g / 125, nt0 = (g - mt * 125) * 5;
  const int m0 = mt * 16;
  v8f acc[5] = {};
#pragma unroll
  for (int kt = 0; kt < 8; ++kt) {
    if (kt < 7)  // uniform branch: prefetch next K-step's B-tile group (L2 -> WGP)
      __builtin_prefetch(Wo_swz + ((size_t)(kt + 1) * 625 + nt0) * 512 + lane * 16, 0, 3);
    v16h a = load_a16(seq_h + (size_t)m0 * HH + kt * 32, HH, lane);
#pragma unroll
    for (int j = 0; j < 5; ++j) {
      v16h b = load_b16(Wo_swz + ((size_t)kt * 625 + nt0 + j) * 512, lane);
      acc[j] = wmma16(a, b, acc[j]);
    }
  }
  const int mBase = m0 + ((lane < 16) ? 0 : 8);
#pragma unroll
  for (int j = 0; j < 5; ++j) {
    const int n = (nt0 + j) * 16 + (lane & 15);
    const float bias = bo[n];
#pragma unroll
    for (int r = 0; r < 8; ++r)
      out[(size_t)(mBase + r) * VV + n] = acc[j][r] + bias;
  }
}

extern "C" void kernel_launch(void* const* d_in, const int* in_sizes, int n_in,
                              void* d_out, int out_size, void* d_ws, size_t ws_size,
                              hipStream_t stream) {
  const int*   words = (const int*)  d_in[0];
  const float* midis = (const float*)d_in[1];
  const float* Emb   = (const float*)d_in[2];
  const float* Wm    = (const float*)d_in[3];
  const float* bm    = (const float*)d_in[4];
  const float* Wx    = (const float*)d_in[5];
  const float* bx    = (const float*)d_in[6];
  const float* U     = (const float*)d_in[7];
  const float* bu    = (const float*)d_in[8];
  const float* Wo    = (const float*)d_in[9];
  const float* bo    = (const float*)d_in[10];
  float* out = (float*)d_out;
  (void)in_sizes; (void)n_in; (void)out_size; (void)ws_size;

  char* ws = (char*)d_ws;
  size_t off = 0;
  auto take = [&](size_t bytes) -> char* {
    char* p = ws + off;
    off += (bytes + 255) & ~(size_t)255;
    return p;
  };
  _Float16* midis_h = (_Float16*)take((size_t)BT * MD * 2);
  _Float16* Wm_swz  = (_Float16*)take((size_t)4  * 20  * 512 * 2);
  _Float16* Wx_swz  = (_Float16*)take((size_t)10 * 48  * 512 * 2);
  _Float16* U_swz   = (_Float16*)take((size_t)8  * 48  * 512 * 2);
  _Float16* Wo_swz  = (_Float16*)take((size_t)8  * 625 * 512 * 2);
  _Float16* x_h     = (_Float16*)take((size_t)BT * EP * 2);
  float*    gx      = (float*)   take((size_t)BT * H3 * 4);
  _Float16* seq_h   = (_Float16*)take((size_t)BT * HH * 2);

  // Stage: convert + swizzle weights into WMMA B-operand layout (L2-resident).
  k_f32_to_f16<<<(BT * MD + 255) / 256, 256, 0, stream>>>(midis, midis_h, BT * MD);
  k_pack_b<<<(40960   + 255) / 256, 256, 0, stream>>>(Wm, Wm_swz, MD, ED, 20,  40960);
  k_pack_b<<<(245760  + 255) / 256, 256, 0, stream>>>(Wx, Wx_swz, ED, H3, 48,  245760);
  k_pack_b<<<(196608  + 255) / 256, 256, 0, stream>>>(U,  U_swz,  HH, H3, 48,  196608);
  k_pack_b<<<(2560000 + 255) / 256, 256, 0, stream>>>(Wo, Wo_swz, HH, VV, 625, 2560000);

  k1_embed<<<320,  256, 0, stream>>>(midis_h, Wm_swz, words, Emb, bm, x_h);
  k2_gx   <<<768,  256, 0, stream>>>(x_h, Wx_swz, bx, gx);
  k3_gru  <<<2,    768, 0, stream>>>(gx, U_swz, bu, words, seq_h);
  k4_out  <<<8000, 256, 0, stream>>>(seq_h, Wo_swz, bo, out);
}
